// SSM_black_42356967473612
// MI455X (gfx1250) — matrix operand order repaired and checked by reading, not compiled, not run on hardware
//
#include <hip/hip_runtime.h>
#include <hip/hip_bf16.h>

// ---------------- types ----------------
typedef __attribute__((ext_vector_type(16))) __bf16 bf16x16;
typedef __attribute__((ext_vector_type(8)))  float  f32x8;
typedef __attribute__((ext_vector_type(4)))  float  f32x4;

#define T_STEPS 1024
#define BSZ     1024
#define NX      128
#define NU      64
#define ND      32
#define NM      32
#define NDT     32
#define NH      128

// ws (bf16 element offsets): pre-packed B-operand tiles [kc][nt][lane=n+16*((k>>3)&1)][elem=(k&7)+8*(k>>4)]
#define WS_WEFF 0        // [4][8][32][16] = 16384
#define WS_HF1  16384    // [2][8][32][16] =  8192
#define WS_HF2  24576    // [4][4][32][16] =  8192
#define WS_BW   32768    // [2][8][32][16] =  8192
#define WS_EW   40960    // [1][8][32][16] =  4096
#define WS_END  45056    // *2 bytes = 90112 B

#define LDSTR 132   // padded f32 row stride (bank-conflict avoidance)

// ---------------- device helpers ----------------
__device__ __forceinline__ f32x8 wmma_bf16(bf16x16 a, bf16x16 b, f32x8 c) {
    return __builtin_amdgcn_wmma_f32_16x16x32_bf16(false, a, false, b, (short)0, c,
                                                   false, false);
}

// A-operand (16x32 bf16) per-lane load from f32 memory.
// lane holds m = lane&15, side = lane>>4; K runs: [side*8, +8) and [side*8+16, +8)
// p0 points at row_base + kchunk*32 + side*8
__device__ __forceinline__ bf16x16 a_from_f32(const float* p0) {
    bf16x16 a;
#pragma unroll
    for (int i = 0; i < 8; i++) a[i]     = (__bf16)p0[i];
#pragma unroll
    for (int i = 0; i < 8; i++) a[i + 8] = (__bf16)p0[16 + i];
    return a;
}

// store a C/D (16x16 f32) accumulator tile into an LDS f32 [16][LDSTR] buffer
__device__ __forceinline__ void store_c(float* wb, int nt, int lane, f32x8 c) {
    int n  = nt * 16 + (lane & 15);
    int mb = (lane >> 4) * 8;
#pragma unroll
    for (int r = 0; r < 8; r++) wb[(mb + r) * LDSTR + n] = c[r];
}

// single-instruction ReLU: force v_max_num_f32 dst, src, 0 (WMMA outputs are
// finite, so skip LLVM's fmaxnum NaN-canonicalize double-max lowering)
__device__ __forceinline__ float relu1(float x) {
    float y;
    asm("v_max_num_f32 %0, %1, 0" : "=v"(y) : "v"(x));
    return y;
}
__device__ __forceinline__ void relu8(f32x8& c) {
#pragma unroll
    for (int r = 0; r < 8; r++) c[r] = relu1(c[r]);
}

// ---------------- kernel 1: W_eff + weight repack (tiny) ----------------
__global__ void prep_kernel(const float* __restrict__ Aw, const float* __restrict__ As,
                            const float* __restrict__ Bw, const float* __restrict__ Ew,
                            const float* __restrict__ hf1, const float* __restrict__ hf2,
                            __bf16* __restrict__ ws) {
    int r = threadIdx.x;            // 0..127 ; this thread owns output column n=r
    int nt = r >> 4, h = r & 15;

    // softmax over row r of A_weight, scaled: gives W_eff[:, r]
    float row[NX];
    float mx = -1e30f;
    for (int c = 0; c < NX; c++) { row[c] = Aw[r * NX + c]; mx = fmaxf(mx, row[c]); }
    float s = 0.0f;
    for (int c = 0; c < NX; c++) { row[c] = __expf(row[c] - mx); s += row[c]; }
    float inv = 1.0f / s;
    for (int c = 0; c < NX; c++) {
        float sig = 1.0f / (1.0f + __expf(-As[r * NX + c]));
        row[c] = (1.0f - 0.1f * sig) * row[c] * inv;   // = W_eff[c][r]
    }
    for (int kc = 0; kc < 4; kc++)
        for (int l = 0; l < 32; l++)
            ws[WS_WEFF + ((kc * 8 + nt) * 32 + h + 16 * ((l >> 3) & 1)) * 16 + (l & 7) + 8 * (l >> 4)] = (__bf16)row[kc * 32 + l];

    // hf1: B[k][n] = hf1_w[n][k], k in [0,64)
    for (int kc = 0; kc < 2; kc++)
        for (int l = 0; l < 32; l++)
            ws[WS_HF1 + ((kc * 8 + nt) * 32 + h + 16 * ((l >> 3) & 1)) * 16 + (l & 7) + 8 * (l >> 4)] = (__bf16)hf1[r * 64 + kc * 32 + l];

    // hf2: B[k][n] = hf2_w[n][k], n in [0,64), k in [0,128)
    if (r < NU) {
        for (int kc = 0; kc < 4; kc++)
            for (int l = 0; l < 32; l++)
                ws[WS_HF2 + ((kc * 4 + nt) * 32 + h + 16 * ((l >> 3) & 1)) * 16 + (l & 7) + 8 * (l >> 4)] = (__bf16)hf2[r * 128 + kc * 32 + l];
    }
    // B_w: B[k][n] = B_w[n][k], k in [0,64)
    for (int kc = 0; kc < 2; kc++)
        for (int l = 0; l < 32; l++)
            ws[WS_BW + ((kc * 8 + nt) * 32 + h + 16 * ((l >> 3) & 1)) * 16 + (l & 7) + 8 * (l >> 4)] = (__bf16)Bw[r * 64 + kc * 32 + l];
    // E_w: B[k][n] = E_w[n][k], k in [0,32)
    for (int l = 0; l < 32; l++)
        ws[WS_EW + (nt * 32 + h + 16 * ((l >> 3) & 1)) * 16 + (l & 7) + 8 * (l >> 4)] = (__bf16)Ew[r * 32 + l];
}

// ---------------- kernel 2: drive = relu(relu(md@hf1^T)@hf2^T)@B_w^T + D@E_w^T ----
// One wave per 16-row tile; drive stored blocked into the X span of d_out:
//   span(g) = Xout + g*2048 ; layout [nt(8)][lane(32)][r(8)] f32
__global__ __launch_bounds__(128) void drive_kernel(const float* __restrict__ Mf,
                                                    const float* __restrict__ DTin,
                                                    const float* __restrict__ Din,
                                                    const __bf16* __restrict__ ws,
                                                    float* __restrict__ Xout) {
    extern __shared__ char smem_raw[];
    __bf16* wB   = (__bf16*)smem_raw;                 // 28672 bf16 = 57344 B
    float*  wbuf = (float*)(smem_raw + 57344);        // 4 waves * 16*LDSTR f32

    // stage packed weights (hf1|hf2|Bw|Ew) from ws into LDS
    {
        const f32x4* src = (const f32x4*)(ws + WS_HF1);
        f32x4*       dst = (f32x4*)wB;
        for (int i = threadIdx.x; i < 3584; i += blockDim.x) dst[i] = src[i];
    }
    __syncthreads();

    const bf16x16* Bv = (const bf16x16*)wB;   // vector-of-16 units
    // vec bases: hf1=0, hf2=512, Bw=1024, Ew=1536

    int lane = threadIdx.x & 31;
    int m = lane & 15, side = lane >> 4;
    float* wb = wbuf + (threadIdx.x >> 5) * (16 * LDSTR);

    long gwave  = (long)((blockIdx.x * blockDim.x + threadIdx.x) >> 5);
    long nwaves = (long)((gridDim.x * blockDim.x) >> 5);
    const long ntiles = (long)T_STEPS * BSZ / 16;

    for (long g = gwave; g < ntiles; g += nwaves) {
        long row0 = g * 16;   // flattened t*BSZ + b
        bf16x16 a0 = a_from_f32(Mf   + (row0 + m) * NM  + side * 8);  // md k 0..31
        bf16x16 a1 = a_from_f32(DTin + (row0 + m) * NDT + side * 8);  // md k 32..63
        bf16x16 aD = a_from_f32(Din  + (row0 + m) * ND  + side * 8);

        // h = relu(md @ hf1^T)  (16 x 128)
#pragma unroll
        for (int nt = 0; nt < 8; nt++) {
            f32x8 c = {};
            c = wmma_bf16(a0, Bv[(0 * 8 + nt) * 32 + lane], c);
            c = wmma_bf16(a1, Bv[(1 * 8 + nt) * 32 + lane], c);
            relu8(c);
            store_c(wb, nt, lane, c);
        }
        bf16x16 ah[4];
#pragma unroll
        for (int kc = 0; kc < 4; kc++)
            ah[kc] = a_from_f32(wb + m * LDSTR + kc * 32 + side * 8);

        // u = relu(h @ hf2^T)   (16 x 64), reuse wave buffer cols 0..63
#pragma unroll
        for (int nt = 0; nt < 4; nt++) {
            f32x8 c = {};
#pragma unroll
            for (int kc = 0; kc < 4; kc++)
                c = wmma_bf16(ah[kc], Bv[512 + (kc * 4 + nt) * 32 + lane], c);
            relu8(c);
            store_c(wb, nt, lane, c);
        }
        bf16x16 au0 = a_from_f32(wb + m * LDSTR + 0  + side * 8);
        bf16x16 au1 = a_from_f32(wb + m * LDSTR + 32 + side * 8);

        // drive = u @ B_w^T + D @ E_w^T   (16 x 128) -> blocked store into X span
#pragma unroll
        for (int nt = 0; nt < 8; nt++) {
            f32x8 c = {};
            c = wmma_bf16(aD,  Bv[1536 + nt * 32 + lane], c);
            c = wmma_bf16(au0, Bv[1024 + (0 * 8 + nt) * 32 + lane], c);
            c = wmma_bf16(au1, Bv[1024 + (1 * 8 + nt) * 32 + lane], c);
            float* dst = Xout + g * 2048 + nt * 256 + lane * 8;
            f32x4 lo = { c[0], c[1], c[2], c[3] };
            f32x4 hi = { c[4], c[5], c[6], c[7] };
            *(f32x4*)dst       = lo;
            *(f32x4*)(dst + 4) = hi;
        }
    }
}

// ---------------- kernel 3: sequential scan  x_t = x_{t-1} @ W_eff + drive_t -------
// 2 waves per 16-batch-row group split the 8 N-tiles; W_eff B-operands held in VGPRs
// for all 1024 steps. State x double-buffered in LDS (f32).
__global__ __launch_bounds__(256) void scan_kernel(const float* __restrict__ x0,
                                                   const __bf16* __restrict__ ws,
                                                   float* __restrict__ Xout,
                                                   float* __restrict__ Yout) {
    extern __shared__ char smem_raw[];
    float* xs = (float*)smem_raw;    // 4 groups * 2 bufs * 16*LDSTR f32

    int tid = threadIdx.x, lane = tid & 31, w = tid >> 5;
    int group = w >> 1, q = w & 1;
    int m = lane & 15, side = lane >> 4;
    int btile = blockIdx.x * 4 + group;       // 16-row batch tile id (0..63)
    long b0 = (long)btile * 16;

    // W_eff B operands for this wave's 4 N-tiles, all K chunks -> registers
    bf16x16 W[4][4];
    const bf16x16* Wv = (const bf16x16*)(ws + WS_WEFF);
#pragma unroll
    for (int kc = 0; kc < 4; kc++)
#pragma unroll
        for (int j = 0; j < 4; j++)
            W[kc][j] = Wv[(kc * 8 + (q * 4 + j)) * 32 + lane];

    float* buf0 = xs + group * 2 * (16 * LDSTR);
    float* buf1 = buf0 + 16 * LDSTR;

    // stage initial x into buf0 (coalesced: lane -> row m, 32-f32 run)
    {
        int row = lane & 15;
        int off = q * 64 + side * 32;
        const float* src = x0 + (b0 + row) * NX + off;
        float* dst = buf0 + row * LDSTR + off;
#pragma unroll
        for (int i = 0; i < 8; i++) ((f32x4*)dst)[i] = ((const f32x4*)src)[i];
    }
    __syncthreads();

    for (int t = 0; t < T_STEPS; t++) {
        float* cur = (t & 1) ? buf1 : buf0;
        float* nxt = (t & 1) ? buf0 : buf1;
        long span = ((long)t * 64 + btile) * 2048;

        // C init = drive tiles (blocked layout in X span): issue ALL loads up front
        // so their latency overlaps the LDS->bf16 A-operand prep below.
        f32x8 acc[4];
#pragma unroll
        for (int j = 0; j < 4; j++) {
            const float* dp = Xout + span + (q * 4 + j) * 256 + lane * 8;
            f32x4 d0 = *(const f32x4*)dp;
            f32x4 d1 = *(const f32x4*)(dp + 4);
#pragma unroll
            for (int i = 0; i < 4; i++) { acc[j][i] = d0[i]; acc[j][i + 4] = d1[i]; }
        }

        // A operands: x tile (16 x 128) from LDS, cvt to bf16
        bf16x16 a[4];
#pragma unroll
        for (int kc = 0; kc < 4; kc++)
            a[kc] = a_from_f32(cur + m * LDSTR + kc * 32 + side * 8);

#pragma unroll
        for (int j = 0; j < 4; j++) {
#pragma unroll
            for (int kc = 0; kc < 4; kc++) acc[j] = wmma_bf16(a[kc], W[kc][j], acc[j]);
            store_c(nxt, q * 4 + j, lane, acc[j]);
        }
        // after this barrier: both waves' x_new is in LDS, and both waves have
        // consumed their drive values into registers -> safe to overwrite span
        __syncthreads();

        // prefetch next step's drive span while we drain this step's stores
        if (t + 1 < T_STEPS) {
            const float* np = Xout + span + 64 * 2048 + (q * 4) * 256 + lane * 8;
#pragma unroll
            for (int j = 0; j < 4; j++) __builtin_prefetch(np + j * 256, 0, 0);
        }

        // X output (row-major) from LDS, fully coalesced 32-f32 runs per lane
        {
            int row = lane & 15;
            int off = q * 64 + side * 32;
            const float* s = nxt + row * LDSTR + off;
            float* d = Xout + span + row * 128 + off;
#pragma unroll
            for (int i = 0; i < 8; i++) ((f32x4*)d)[i] = ((const f32x4*)s)[i];
        }
        if (q == 1 && lane < 16)
            Yout[(long)t * BSZ + b0 + lane] = nxt[lane * LDSTR + 127];
    }
}

// ---------------- launch ----------------
extern "C" void kernel_launch(void* const* d_in, const int* in_sizes, int n_in,
                              void* d_out, int out_size, void* d_ws, size_t ws_size,
                              hipStream_t stream) {
    const float* x   = (const float*)d_in[0];
    const float* Mf  = (const float*)d_in[1];
    const float* DTin= (const float*)d_in[2];
    const float* Din = (const float*)d_in[3];
    const float* Aw  = (const float*)d_in[4];
    const float* As  = (const float*)d_in[5];
    const float* Bw  = (const float*)d_in[6];
    const float* Ew  = (const float*)d_in[7];
    const float* hf1 = (const float*)d_in[8];
    const float* hf2 = (const float*)d_in[9];

    float* Xout = (float*)d_out;
    float* Yout = Xout + (long)T_STEPS * BSZ * NX;
    __bf16* wsb = (__bf16*)d_ws;

    prep_kernel<<<1, 128, 0, stream>>>(Aw, As, Bw, Ew, hf1, hf2, wsb);

    size_t smem_drive = 57344 + 4 * 16 * LDSTR * sizeof(float);   // 91,136 B
    drive_kernel<<<1024, 128, smem_drive, stream>>>(Mf, DTin, Din, wsb, Xout);

    size_t smem_scan = 4 * 2 * 16 * LDSTR * sizeof(float);        // 67,584 B
    scan_kernel<<<16, 256, smem_scan, stream>>>(x, wsb, Xout, Yout);
}
